// BaselineTransformer_12584254178033
// MI455X (gfx1250) — compile-verified, matrix operand order checked by reading, not run on hardware
//
#include <hip/hip_runtime.h>
#include <hip/hip_bf16.h>
#include <math.h>

typedef unsigned short u16;
typedef __bf16 bf16_t;
typedef float v4f  __attribute__((ext_vector_type(4)));
typedef float v8f  __attribute__((ext_vector_type(8)));
typedef bf16_t v16bf __attribute__((ext_vector_type(16)));

#define B_   2
#define T_   1024
#define E_   768
#define NH_  12
#define HD_  64
#define L_   4
#define FF_  3072
#define V_   50257
#define MTOT (B_ * T_)

#define BM 128
#define BN 128
#define BK 32
#define LDSS 40   // ushort stride (80 bytes) -> conflict-free ds_load_b128

#if defined(__HIP_DEVICE_COMPILE__) && \
    __has_builtin(__builtin_amdgcn_global_load_async_to_lds_b128) && \
    __has_builtin(__builtin_amdgcn_s_wait_asynccnt)
#define HAVE_ASYNC_LDS 1
#else
#define HAVE_ASYNC_LDS 0
#endif

#if HAVE_ASYNC_LDS
// builtin prototype (probe-verified on this toolchain):
//   void (v4i addrspace(1)* src, v4i addrspace(3)* dst, imm int off, imm int cpol)
typedef int v4i_vs __attribute__((vector_size(16)));
#define ASYNC_CP16(dst, src)                                              \
    __builtin_amdgcn_global_load_async_to_lds_b128(                       \
        (__attribute__((address_space(1))) v4i_vs*)(void*)(src),          \
        (__attribute__((address_space(3))) v4i_vs*)(void*)(dst), 0, 0)
#endif

struct __align__(16) U4 { unsigned x, y, z, w; };

__device__ __forceinline__ u16 f2bf(float f) {
    unsigned u = __float_as_uint(f);
    u += 0x7fffu + ((u >> 16) & 1u);   // round-to-nearest-even
    return (u16)(u >> 16);
}

// ---------------------------------------------------------------- converts
__global__ void f32_to_bf16_kernel(const float* __restrict__ in,
                                   u16* __restrict__ out, long long n) {
    long long i = (long long)blockIdx.x * 256 + threadIdx.x;
    long long stride = (long long)gridDim.x * 256;
    for (; i < n; i += stride) out[i] = f2bf(in[i]);
}

// ---------------------------------------------------------------- embedding
__global__ void embed_kernel(const int* __restrict__ ids,
                             const float* __restrict__ tok,
                             const float* __restrict__ pos,
                             float* __restrict__ x) {
    int row = blockIdx.x;                 // 0 .. MTOT-1
    int t = row & (T_ - 1);
    long long id = ids[row];
    const float* te = tok + id * (long long)E_;
    const float* pe = pos + (long long)t * E_;
    float* xr = x + (long long)row * E_;
    for (int e = threadIdx.x; e < E_; e += 256) xr[e] = te[e] + pe[e];
}

// ---------------------------------------------------------------- layernorm
__global__ void layernorm_kernel(const float* __restrict__ x,
                                 const float* __restrict__ sc,
                                 const float* __restrict__ bi,
                                 u16* __restrict__ out) {
    __shared__ float2 rb[256];
    int row = blockIdx.x;
    int tid = threadIdx.x;
    const float* xr = x + (long long)row * E_;
    float v[3];
    float s = 0.f, ss = 0.f;
#pragma unroll
    for (int i = 0; i < 3; ++i) {
        v[i] = xr[tid + i * 256];
        s += v[i]; ss += v[i] * v[i];
    }
    rb[tid] = make_float2(s, ss);
    __syncthreads();
    for (int o = 128; o > 0; o >>= 1) {
        if (tid < o) {
            rb[tid].x += rb[tid + o].x;
            rb[tid].y += rb[tid + o].y;
        }
        __syncthreads();
    }
    float mean = rb[0].x * (1.f / E_);
    float var  = rb[0].y * (1.f / E_) - mean * mean;
    float rstd = rsqrtf(var + 1e-5f);
    u16* orow = out + (long long)row * E_;
#pragma unroll
    for (int i = 0; i < 3; ++i) {
        int e = tid + i * 256;
        orow[e] = f2bf((v[i] - mean) * rstd * sc[e] + bi[e]);
    }
}

// ---------------------------------------------------------------- softmax
__global__ void softmax_causal_kernel(const float* __restrict__ S,
                                      u16* __restrict__ P) {
    __shared__ float rb[256];
    long long row = blockIdx.x;               // bh*T + tq
    int tq = (int)(row & (T_ - 1));
    int tid = threadIdx.x;
    const float* s = S + row * T_;
    u16* p = P + row * T_;
    float v[4];
    float m = -3.0e38f;
#pragma unroll
    for (int i = 0; i < 4; ++i) {
        int tk = tid + i * 256;
        float val = (tk <= tq) ? s[tk] : -3.0e38f;
        v[i] = val; m = fmaxf(m, val);
    }
    rb[tid] = m; __syncthreads();
    for (int o = 128; o > 0; o >>= 1) {
        if (tid < o) rb[tid] = fmaxf(rb[tid], rb[tid + o]);
        __syncthreads();
    }
    m = rb[0];
    __syncthreads();
    float sum = 0.f;
#pragma unroll
    for (int i = 0; i < 4; ++i) {
        int tk = tid + i * 256;
        float e = (tk <= tq) ? __expf(v[i] - m) : 0.f;
        v[i] = e; sum += e;
    }
    rb[tid] = sum; __syncthreads();
    for (int o = 128; o > 0; o >>= 1) {
        if (tid < o) rb[tid] += rb[tid + o];
        __syncthreads();
    }
    float inv = 1.f / rb[0];
#pragma unroll
    for (int i = 0; i < 4; ++i) p[tid + i * 256] = f2bf(v[i] * inv);
}

// ---------------------------------------------------------------- WMMA GEMM
// C[M,N] = A[M,K] (bf16, row major, lda) x W[N,K]^T (bf16, row major, ldw)
// FULLN=true: the N-tile is guaranteed fully in range -> branchless async path
enum { EPI_F32 = 0, EPI_BF16 = 1, EPI_QKV = 2, EPI_GELU = 3, EPI_RES = 4 };

union Frag { v4f f4[2]; v16bf bf; };

template <int EPI, bool FULLN>
__global__ __launch_bounds__(256)
void gemm_bf16_wmma(const u16* __restrict__ A, int lda, long long sA,
                    const u16* __restrict__ W, int ldw, long long sW,
                    void* __restrict__ Cp, int ldc,
                    long long sCo, long long sCi, int innerCnt,
                    const float* __restrict__ bias,
                    int M, int N, int K, int tileNbase,
                    u16* __restrict__ qo, u16* __restrict__ ko,
                    u16* __restrict__ vo) {
    __shared__ u16 lsA[2][BM * LDSS];
    __shared__ u16 lsW[2][BN * LDSS];

    int z = blockIdx.z;
    A += (long long)z * sA;
    W += (long long)z * sW;
    long long coff = (long long)(z / innerCnt) * sCo +
                     (long long)(z % innerCnt) * sCi;

    const int tileN = tileNbase + blockIdx.x * BN;
    const int tileM = blockIdx.y * BM;
    const int tid = threadIdx.x;
    const int lane = tid & 31;
    const int wave = tid >> 5;
    const int half = lane >> 4;        // 0: lanes 0-15, 1: lanes 16-31
    const int lm = lane & 15;
    const int klo = half * 8;          // K sub-chunk base (ushort units)

    v8f acc[8];
#pragma unroll
    for (int nt = 0; nt < 8; ++nt)
#pragma unroll
        for (int r = 0; r < 8; ++r) acc[nt][r] = 0.f;

    // cooperative tile staging: 256 thr, one 32-byte strip each per tile
    const int ldr  = tid >> 1;             // 0..127
    const int ldc0 = (tid & 1) * 16;       // 0 or 16 ushorts
    const u16* gA = A + (long long)(tileM + ldr) * lda + ldc0;
    const u16* gW = W + (long long)(tileN + ldr) * ldw + ldc0;
    const bool wIn = FULLN || ((tileN + ldr) < N);
    const int sOff = ldr * LDSS + ldc0;
    const int arow = wave * 16 + lm;
    const int nK = K / BK;

    U4 ra0, ra1, rw0, rw1;

    // compute one 16(M)x128(N) strip from staged buffer p
    auto computeTile = [&](int p) {
        Frag fa;
        {
            const u16* la = &lsA[p][arow * LDSS];
            fa.f4[0] = *(const v4f*)(la + klo);
            fa.f4[1] = *(const v4f*)(la + klo + 16);
        }
        Frag fb[2];
        {
            const u16* lw = &lsW[p][lm * LDSS];
            fb[0].f4[0] = *(const v4f*)(lw + klo);
            fb[0].f4[1] = *(const v4f*)(lw + klo + 16);
        }
#pragma unroll
        for (int nt = 0; nt < 8; ++nt) {
            if (nt + 1 < 8) {
                const u16* lw = &lsW[p][((nt + 1) * 16 + lm) * LDSS];
                fb[(nt + 1) & 1].f4[0] = *(const v4f*)(lw + klo);
                fb[(nt + 1) & 1].f4[1] = *(const v4f*)(lw + klo + 16);
            }
            acc[nt] = __builtin_amdgcn_wmma_f32_16x16x32_bf16(
                false, fa.bf, false, fb[nt & 1].bf, (short)0, acc[nt],
                false, false);
        }
    };

    // ---- stage tile 0 into buffer 0
#if HAVE_ASYNC_LDS
    if constexpr (FULLN) {
        ASYNC_CP16(&lsA[0][sOff],     gA);
        ASYNC_CP16(&lsA[0][sOff + 8], gA + 8);
        ASYNC_CP16(&lsW[0][sOff],     gW);
        ASYNC_CP16(&lsW[0][sOff + 8], gW + 8);
        __builtin_amdgcn_s_wait_asynccnt(0);
    } else
#endif
    {
        ra0 = *(const U4*)(gA);
        ra1 = *(const U4*)(gA + 8);
        rw0 = {0, 0, 0, 0}; rw1 = {0, 0, 0, 0};
        if (wIn) { rw0 = *(const U4*)(gW); rw1 = *(const U4*)(gW + 8); }
        *(U4*)&lsA[0][sOff]     = ra0;
        *(U4*)&lsA[0][sOff + 8] = ra1;
        *(U4*)&lsW[0][sOff]     = rw0;
        *(U4*)&lsW[0][sOff + 8] = rw1;
    }
    __syncthreads();

    // ---- steady state: stage(i+1) | compute(i) | land | barrier
    for (int i = 0; i < nK - 1; ++i) {
        const int p = i & 1;
        const u16* gAn = gA + (long long)(i + 1) * BK;
        const u16* gWn = gW + (long long)(i + 1) * BK;

#if HAVE_ASYNC_LDS
        if constexpr (FULLN) {
            ASYNC_CP16(&lsA[p ^ 1][sOff],     gAn);
            ASYNC_CP16(&lsA[p ^ 1][sOff + 8], gAn + 8);
            ASYNC_CP16(&lsW[p ^ 1][sOff],     gWn);
            ASYNC_CP16(&lsW[p ^ 1][sOff + 8], gWn + 8);
        } else
#endif
        {
            ra0 = *(const U4*)(gAn);
            ra1 = *(const U4*)(gAn + 8);
            rw0 = {0, 0, 0, 0}; rw1 = {0, 0, 0, 0};
            if (wIn) { rw0 = *(const U4*)(gWn); rw1 = *(const U4*)(gWn + 8); }
        }

        computeTile(p);

#if HAVE_ASYNC_LDS
        if constexpr (FULLN) {
            __builtin_amdgcn_s_wait_asynccnt(0);
        } else
#endif
        {
            *(U4*)&lsA[p ^ 1][sOff]     = ra0;
            *(U4*)&lsA[p ^ 1][sOff + 8] = ra1;
            *(U4*)&lsW[p ^ 1][sOff]     = rw0;
            *(U4*)&lsW[p ^ 1][sOff + 8] = rw1;
        }
        __syncthreads();
    }
    computeTile((nK - 1) & 1);

    // ------------------------------------------------------------ epilogue
    const int rowBase = tileM + wave * 16 + half * 8;
#pragma unroll
    for (int nt = 0; nt < 8; ++nt) {
        const int col = tileN + nt * 16 + lm;
        if constexpr (!FULLN) { if (col >= N) continue; }
#pragma unroll
        for (int r = 0; r < 8; ++r) {
            const int row = rowBase + r;
            float c = acc[nt][r];
            if constexpr (EPI == EPI_F32) {
                ((float*)Cp + coff)[(long long)row * ldc + col] = c;
            } else if constexpr (EPI == EPI_BF16) {
                ((u16*)Cp + coff)[(long long)row * ldc + col] = f2bf(c);
            } else if constexpr (EPI == EPI_GELU) {
                c += bias[col];
                c = 0.5f * c * (1.f + erff(c * 0.70710678118f));
                ((u16*)Cp + coff)[(long long)row * ldc + col] = f2bf(c);
            } else if constexpr (EPI == EPI_RES) {
                float bv = (bias != nullptr) ? bias[col] : 0.f;
                float* X = (float*)Cp + coff;
                long long idx = (long long)row * ldc + col;
                X[idx] = X[idx] + c + bv;
            } else {  // EPI_QKV
                const int b = row >> 10, t = row & (T_ - 1);
                const int region = col / E_;
                const int cc = col - region * E_;
                const int h = cc >> 6, hd = cc & (HD_ - 1);
                const long long bh = (long long)(b * NH_ + h);
                if (region == 0)
                    qo[(bh * T_ + t) * HD_ + hd] = f2bf(c * 0.125f);  // HD^-0.5
                else if (region == 1)
                    ko[(bh * T_ + t) * HD_ + hd] = f2bf(c);
                else
                    vo[(bh * HD_ + hd) * T_ + t] = f2bf(c);           // V^T
            }
        }
    }
}

// ---------------------------------------------------------------- launcher
extern "C" void kernel_launch(void* const* d_in, const int* in_sizes, int n_in,
                              void* d_out, int out_size, void* d_ws, size_t ws_size,
                              hipStream_t stream) {
    const int*   ids  = (const int*)d_in[0];
    const float* tok  = (const float*)d_in[1];
    const float* pos  = (const float*)d_in[2];
    const float* ln1s = (const float*)d_in[3];
    const float* ln1b = (const float*)d_in[4];
    const float* qkvw = (const float*)d_in[5];
    const float* outw = (const float*)d_in[6];
    const float* ln2s = (const float*)d_in[7];
    const float* ln2b = (const float*)d_in[8];
    const float* fc1w = (const float*)d_in[9];
    const float* fc1b = (const float*)d_in[10];
    const float* fc2w = (const float*)d_in[11];
    const float* fc2b = (const float*)d_in[12];
    const float* lnfs = (const float*)d_in[13];
    const float* lnfb = (const float*)d_in[14];

    size_t cur = 0;
    auto take = [&](size_t bytes) -> char* {
        cur = (cur + 255) & ~(size_t)255;
        char* p = (char*)d_ws + cur;
        cur += bytes;
        return p;
    };

    float* x    = (float*)take((size_t)MTOT * E_ * 4);
    u16*   h    = (u16*)  take((size_t)MTOT * E_ * 2);
    u16*   q    = (u16*)  take((size_t)B_ * NH_ * T_ * HD_ * 2);
    u16*   kb   = (u16*)  take((size_t)B_ * NH_ * T_ * HD_ * 2);
    u16*   vt   = (u16*)  take((size_t)B_ * NH_ * T_ * HD_ * 2);
    float* S    = (float*)take((size_t)B_ * NH_ * T_ * T_ * 4);
    u16*   P    = (u16*)  take((size_t)B_ * NH_ * T_ * T_ * 2);
    u16*   ao   = (u16*)  take((size_t)MTOT * E_ * 2);
    u16*   ffh  = (u16*)  take((size_t)MTOT * FF_ * 2);
    u16*   wqkv = (u16*)  take((size_t)L_ * 3 * E_ * E_ * 2);
    u16*   wout = (u16*)  take((size_t)L_ * E_ * E_ * 2);
    u16*   wfc1 = (u16*)  take((size_t)L_ * FF_ * E_ * 2);
    u16*   wfc2 = (u16*)  take((size_t)L_ * E_ * FF_ * 2);
    u16*   wtok = (u16*)  take((size_t)V_ * E_ * 2);

    // --- weight conversion f32 -> bf16 ---
    dim3 cg(4096), cb(256);
    f32_to_bf16_kernel<<<cg, cb, 0, stream>>>(qkvw, wqkv, (long long)L_ * 3 * E_ * E_);
    f32_to_bf16_kernel<<<cg, cb, 0, stream>>>(outw, wout, (long long)L_ * E_ * E_);
    f32_to_bf16_kernel<<<cg, cb, 0, stream>>>(fc1w, wfc1, (long long)L_ * FF_ * E_);
    f32_to_bf16_kernel<<<cg, cb, 0, stream>>>(fc2w, wfc2, (long long)L_ * E_ * FF_);
    f32_to_bf16_kernel<<<cg, cb, 0, stream>>>(tok,  wtok, (long long)V_ * E_);

    // --- embedding ---
    embed_kernel<<<dim3(MTOT), dim3(256), 0, stream>>>(ids, tok, pos, x);

    const long long sQKh = (long long)T_ * HD_;  // per-(b,h) q/k stride
    for (int l = 0; l < L_; ++l) {
        // LN1
        layernorm_kernel<<<dim3(MTOT), dim3(256), 0, stream>>>(
            x, ln1s + l * E_, ln1b + l * E_, h);
        // QKV projection: M=2048, N=2304, K=768
        gemm_bf16_wmma<EPI_QKV, true><<<dim3(3 * E_ / BN, MTOT / BM, 1), dim3(256), 0, stream>>>(
            h, E_, 0, wqkv + (size_t)l * 3 * E_ * E_, E_, 0,
            nullptr, 0, 0, 0, 1, nullptr, MTOT, 3 * E_, E_, 0, q, kb, vt);
        // scores: per (b,h): S = Q K^T  (M=N=1024, K=64), scale folded into Q
        gemm_bf16_wmma<EPI_F32, true><<<dim3(T_ / BN, T_ / BM, B_ * NH_), dim3(256), 0, stream>>>(
            q, HD_, sQKh, kb, HD_, sQKh,
            S, T_, (long long)T_ * T_, 0, 1, nullptr, T_, T_, HD_, 0,
            nullptr, nullptr, nullptr);
        // causal softmax -> P (bf16)
        softmax_causal_kernel<<<dim3(B_ * NH_ * T_), dim3(256), 0, stream>>>(S, P);
        // O = P V: per (b,h): M=1024, N=64, K=1024; out -> ao[b,t,h*64+hd]
        gemm_bf16_wmma<EPI_BF16, false><<<dim3(1, T_ / BM, B_ * NH_), dim3(256), 0, stream>>>(
            P, T_, (long long)T_ * T_, vt, T_, (long long)HD_ * T_,
            ao, E_, (long long)T_ * E_, HD_, NH_, nullptr, T_, HD_, T_, 0,
            nullptr, nullptr, nullptr);
        // out projection + residual: x += ao @ out_w^T
        gemm_bf16_wmma<EPI_RES, true><<<dim3(E_ / BN, MTOT / BM, 1), dim3(256), 0, stream>>>(
            ao, E_, 0, wout + (size_t)l * E_ * E_, E_, 0,
            x, E_, 0, 0, 1, nullptr, MTOT, E_, E_, 0, nullptr, nullptr, nullptr);
        // LN2
        layernorm_kernel<<<dim3(MTOT), dim3(256), 0, stream>>>(
            x, ln2s + l * E_, ln2b + l * E_, h);
        // FFN up + GELU
        gemm_bf16_wmma<EPI_GELU, true><<<dim3(FF_ / BN, MTOT / BM, 1), dim3(256), 0, stream>>>(
            h, E_, 0, wfc1 + (size_t)l * FF_ * E_, E_, 0,
            ffh, FF_, 0, 0, 1, fc1b + l * FF_, MTOT, FF_, E_, 0,
            nullptr, nullptr, nullptr);
        // FFN down + bias + residual
        gemm_bf16_wmma<EPI_RES, true><<<dim3(E_ / BN, MTOT / BM, 1), dim3(256), 0, stream>>>(
            ffh, FF_, 0, wfc2 + (size_t)l * E_ * FF_, FF_, 0,
            x, E_, 0, 0, 1, fc2b + l * E_, MTOT, E_, FF_, 0,
            nullptr, nullptr, nullptr);
    }

    // final LN
    layernorm_kernel<<<dim3(MTOT), dim3(256), 0, stream>>>(x, lnfs, lnfb, h);
    // logits = h @ tok_emb^T : M=2048, N=50257, K=768
    // bulk: 392 full N-tiles (async path), tail: 1 guarded tile
    const int nFull = V_ / BN;                 // 392
    gemm_bf16_wmma<EPI_F32, true><<<dim3(nFull, MTOT / BM, 1), dim3(256), 0, stream>>>(
        h, E_, 0, wtok, E_, 0,
        d_out, V_, 0, 0, 1, nullptr, MTOT, V_, E_, 0,
        nullptr, nullptr, nullptr);
    gemm_bf16_wmma<EPI_F32, false><<<dim3(1, MTOT / BM, 1), dim3(256), 0, stream>>>(
        h, E_, 0, wtok, E_, 0,
        d_out, V_, 0, 0, 1, nullptr, MTOT, V_, E_, nFull * BN,
        nullptr, nullptr, nullptr);
}
